// GatedConvParityNetwork_76982993813507
// MI455X (gfx1250) — matrix operand order repaired and checked by reading, not compiled, not run on hardware
//
#include <hip/hip_runtime.h>

// ---------------------------------------------------------------------------
// GatedConvParityNetwork for MI455X (gfx1250).
// Dominant cost: per-pair radial MLPs (3->100->100->100->nw) over 18432 pairs
// => batched GEMM, done with v_wmma_f32_16x16x32_f16 (f16 in, f32 accum).
// Everything else (pairwise geometry, tensor-product conv, gating) is small
// VALU work. Radial weight tensor w stored f16 (max 113 MB) in workspace.
// ---------------------------------------------------------------------------

typedef __attribute__((ext_vector_type(16))) _Float16 v16h;
typedef __attribute__((ext_vector_type(8)))  float    v8f;
typedef __attribute__((ext_vector_type(4)))  float    f32x4;

#define NB_   2
#define NPTS  96
#define PAIRS (NB_*NPTS*NPTS)   // 18432, divisible by 16
#define KP    128               // padded K for radial MLP GEMMs
#define HID   100

__device__ __forceinline__ float silu_f(float x)   { return x * (1.0f / (1.0f + __expf(-x))); }
__device__ __forceinline__ float sigm_f(float x)   { return 1.0f / (1.0f + __expf(-x)); }

// ------------------------- layer spec tables -------------------------------
struct RsEnt   { int m, l, off; };            // multiplicity, l, feature offset
struct PathEnt { int io, ii, lf, woff, np; }; // np = npaths[io]

__constant__ int c_nRsIn[4]  = {1, 2, 3, 4};
__constant__ int c_nRsOut[4] = {3, 4, 5, 1};
__constant__ int c_nPaths[4] = {3, 7, 10, 2};
__constant__ int c_din[4]    = {4, 64, 112, 128};
__constant__ int c_dout[4]   = {80, 144, 160, 8};

__constant__ RsEnt c_RsIn[4][4] = {
  { {4,0,0},   {0,0,0},   {0,0,0},   {0,0,0}  },
  { {16,0,0},  {16,1,16}, {0,0,0},   {0,0,0}  },
  { {16,0,0},  {16,1,16}, {16,1,64}, {0,0,0}  },
  { {16,0,0},  {16,0,16}, {16,1,32}, {16,1,80}},
};
__constant__ RsEnt c_RsOut[4][5] = {
  { {16,0,0}, {16,0,16}, {16,1,32}, {0,0,0},   {0,0,0}   },
  { {16,0,0}, {32,0,16}, {16,1,48}, {16,1,96}, {0,0,0}   },
  { {16,0,0}, {16,0,16}, {32,0,32}, {16,1,64}, {16,1,112}},
  { {8,0,0},  {0,0,0},   {0,0,0},   {0,0,0},   {0,0,0}   },
};
__constant__ PathEnt c_Paths[4][10] = {
  { {0,0,0,0,1},{1,0,0,64,1},{2,0,1,128,1},
    {0,0,0,0,0},{0,0,0,0,0},{0,0,0,0,0},{0,0,0,0,0},{0,0,0,0,0},{0,0,0,0,0},{0,0,0,0,0} },
  { {0,0,0,0,2},{0,1,1,256,2},{1,0,0,512,2},{1,1,1,1024,2},{2,1,1,1536,1},
    {3,0,1,1792,2},{3,1,0,2048,2},
    {0,0,0,0,0},{0,0,0,0,0},{0,0,0,0,0} },
  { {0,0,0,0,2},{0,2,1,256,2},{1,1,1,512,1},{2,0,0,768,2},{2,2,1,1280,2},
    {3,1,0,1792,2},{3,2,1,2048,2},{4,0,1,2304,3},{4,1,1,2560,3},{4,2,0,2816,3} },
  { {0,0,0,0,2},{0,3,1,128,2},
    {0,0,0,0,0},{0,0,0,0,0},{0,0,0,0,0},{0,0,0,0,0},{0,0,0,0,0},{0,0,0,0,0},{0,0,0,0,0},{0,0,0,0,0} },
};

// gate tables (layers 0..2)
__constant__ int c_gNscalBlk[3] = {1, 1, 2};
__constant__ int c_gPar[3][2]   = {{1,0},{1,0},{1,-1}};
__constant__ int c_gNgates[3]   = {16, 32, 32};
__constant__ int c_gDout[3]     = {64, 112, 128};

// ---------------------------------------------------------------------------
// Pairwise geometry: distances, basis (3 gaussians), Y1 = sqrt(3)*unit_vec
// ---------------------------------------------------------------------------
__global__ void pair_kernel(const float* __restrict__ geo,
                            float* __restrict__ Y1, float* __restrict__ basis) {
  int p = blockIdx.x * blockDim.x + threadIdx.x;
  if (p >= PAIRS) return;
  int b   = p / (NPTS * NPTS);
  int rem = p - b * NPTS * NPTS;
  int a = rem / NPTS, n = rem % NPTS;
  const float* ga = geo + (size_t)(b * NPTS + a) * 3;
  const float* gn = geo + (size_t)(b * NPTS + n) * 3;
  float rx = ga[0] - gn[0], ry = ga[1] - gn[1], rz = ga[2] - gn[2];
  float d  = sqrtf(rx*rx + ry*ry + rz*rz + 1e-12f);
  float iv = 1.0f / d;
  const float S3 = 1.7320508075688772f;
  Y1[p*3+0] = S3 * rx * iv;
  Y1[p*3+1] = S3 * ry * iv;
  Y1[p*3+2] = S3 * rz * iv;
  // centers linspace(0,1,3) = {0,0.5,1}; sigma = 0.5 -> ((d-c)/0.5)^2
  float t0 = (d - 0.0f) * 2.0f, t1 = (d - 0.5f) * 2.0f, t2 = (d - 1.0f) * 2.0f;
  basis[p*3+0] = __expf(-t0*t0);
  basis[p*3+1] = __expf(-t1*t1);
  basis[p*3+2] = __expf(-t2*t2);
}

// ---------------------------------------------------------------------------
// First MLP layer (K=3, too thin for WMMA): H1 = silu(basis @ W1 + b1), f16 out
// ---------------------------------------------------------------------------
__global__ void radial1_kernel(const float* __restrict__ basis,
                               const float* __restrict__ W,   // (3,100) row-major
                               const float* __restrict__ bias,
                               _Float16* __restrict__ H) {
  int row = blockIdx.x * blockDim.x + threadIdx.x;
  if (row >= PAIRS) return;
  float x0 = basis[row*3+0], x1 = basis[row*3+1], x2 = basis[row*3+2];
  _Float16* h = H + (size_t)row * KP;
  for (int j = 0; j < HID; ++j) {
    float v = x0 * W[j] + x1 * W[HID + j] + x2 * W[2*HID + j] + bias[j];
    h[j] = (_Float16)silu_f(v);
  }
  for (int j = HID; j < KP; ++j) h[j] = (_Float16)0.0f;
}

// ---------------------------------------------------------------------------
// Weight conversion: W (K x N f32, row-major) -> Wt (Npad x KP f16), Wt[n][k]
// Zero padding in both dims so padded GEMM K/N tiles contribute exactly 0.
// ---------------------------------------------------------------------------
__global__ void convert_wt_kernel(const float* __restrict__ W, int K, int N,
                                  int Npad, _Float16* __restrict__ Wt) {
  int idx = blockIdx.x * blockDim.x + threadIdx.x;
  if (idx >= Npad * KP) return;
  int nr = idx >> 7, k = idx & (KP - 1);
  float v = (nr < N && k < K) ? W[(size_t)k * N + nr] : 0.0f;
  Wt[idx] = (_Float16)v;
}

// ---------------------------------------------------------------------------
// WMMA GEMM: Out = act(A @ Wt^T + bias).  A: (PAIRS x KP f16), Wt: (N x KP f16)
// One 16x16 output tile per wave, K loop of 16x16x32 WMMA steps.
// A-frag per ISA 16-bit A layout: lane(half,r): halves 0..7 = K[k0+8h..],
// halves 8..15 = K[k0+16+8h..] of row r.  B-frag: same K order as A
// of column r (column = row of pre-transposed Wt). C/D: c[j] = C[8h+j][r].
// ---------------------------------------------------------------------------
__global__ void wmma_gemm_kernel(const _Float16* __restrict__ A,
                                 const _Float16* __restrict__ Wt,
                                 const float* __restrict__ bias,
                                 int Mtiles, int Ntiles, int Ncols, int Ksteps,
                                 _Float16* __restrict__ Out, int ldo, int act) {
  int wave = threadIdx.x >> 5;
  int lane = threadIdx.x & 31;
  int half = lane >> 4, r = lane & 15;
  int tile = blockIdx.x * (blockDim.x >> 5) + wave;
  if (tile >= Mtiles * Ntiles) return;      // wave-uniform: EXEC all-ones inside
  int mt = tile / Ntiles, nt = tile % Ntiles;

  v8f acc = {};
  const _Float16* arow = A  + (size_t)(mt * 16 + r) * KP;
  const _Float16* brow = Wt + (size_t)(nt * 16 + r) * KP;
  for (int ks = 0; ks < Ksteps; ++ks) {
    int k0 = ks * 32;
    union { v16h v; f32x4 f[2]; } af, bf;
    const f32x4* ap = (const f32x4*)(arow + k0);
    af.f[0] = ap[half];           // K = k0 + 8*half .. +7
    af.f[1] = ap[2 + half];       // K = k0 + 16 + 8*half .. +7
    const f32x4* bp = (const f32x4*)(brow + k0);
    bf.f[0] = bp[half];           // K = k0 + 8*half .. +7
    bf.f[1] = bp[2 + half];       // K = k0 + 16 + 8*half .. +7
    acc = __builtin_amdgcn_wmma_f32_16x16x32_f16(false, af.v, false, bf.v,
                                                 (short)0, acc, false, false);
  }
  int col = nt * 16 + r;
  float bb = (col < Ncols) ? bias[col] : 0.0f;
  for (int j = 0; j < 8; ++j) {
    float v = acc[j] + bb;
    if (act) v = silu_f(v);
    Out[(size_t)(mt * 16 + half * 8 + j) * ldo + col] = (_Float16)v;
  }
}

// ---------------------------------------------------------------------------
// Tensor-product conv: one block per output point (b,a); neighbor features +
// Y cached in LDS; each thread owns one output component (io,u,oc) and loops
// neighbors x paths, reading R slices from the f16 radial-weight tensor.
// ---------------------------------------------------------------------------
__global__ __launch_bounds__(160)
void conv_kernel(int layer, const float* __restrict__ feat,
                 const _Float16* __restrict__ wbuf, int ldw,
                 const float* __restrict__ Y1, float* __restrict__ out) {
  __shared__ float sfeat[NPTS * 128];
  __shared__ float sy[NPTS * 3];
  int blk = blockIdx.x;
  int b = blk / NPTS, a = blk % NPTS;
  int din = c_din[layer], dout = c_dout[layer];
  int t = threadIdx.x;
  for (int i = t; i < NPTS * din; i += blockDim.x)
    sfeat[i] = feat[(size_t)b * NPTS * din + i];
  for (int i = t; i < NPTS * 3; i += blockDim.x)
    sy[i] = Y1[(size_t)(b * NPTS + a) * NPTS * 3 + i];
  __syncthreads();
  if (t >= dout) return;

  int io = 0, u = 0, oc = 0, lo = 0;
  for (int rIdx = 0; rIdx < c_nRsOut[layer]; ++rIdx) {
    RsEnt e = c_RsOut[layer][rIdx];
    int dsz = e.m * (2 * e.l + 1);
    if (t >= e.off && t < e.off + dsz) {
      io = rIdx; lo = e.l;
      int rel = t - e.off;
      u = rel / (2 * e.l + 1);
      oc = rel % (2 * e.l + 1);
      break;
    }
  }

  const float IS3 = 0.5773502691896258f;   // 1/sqrt(3)
  const float IS6 = 0.4082482904638630f;   // 1/sqrt(6)
  const float RS96 = 0.1020620726159658f;  // 1/sqrt(96)
  size_t pairbase = (size_t)(b * NPTS + a) * NPTS;
  float acc = 0.0f;

  for (int pi = 0; pi < c_nPaths[layer]; ++pi) {
    PathEnt pe = c_Paths[layer][pi];
    if (pe.io != io) continue;
    RsEnt ein = c_RsIn[layer][pe.ii];
    int mi = ein.m, li = ein.l, inoff = ein.off;
    float norm = rsqrtf((float)(mi * pe.np)) * RS96;
    const _Float16* wb = wbuf + pairbase * ldw + pe.woff + u * mi;
    float pacc = 0.0f;
    for (int n = 0; n < NPTS; ++n) {
      const float* f = sfeat + n * din + inoff;
      const _Float16* R = wb + (size_t)n * ldw;
      if (lo == 0 && li == 0) {                 // lf=0: s2[v] = f[v]
        for (int v = 0; v < mi; ++v) pacc += (float)R[v] * f[v];
      } else if (lo == 0 && li == 1) {          // lf=1: s2[v] = (Y . f_v)/s3
        float yx = sy[n*3], yy = sy[n*3+1], yz = sy[n*3+2];
        for (int v = 0; v < mi; ++v) {
          float s = (yx * f[3*v] + yy * f[3*v+1] + yz * f[3*v+2]) * IS3;
          pacc += (float)R[v] * s;
        }
      } else if (lo == 1 && li == 0) {          // lf=1: s2[o,v] = Y[o]*f[v]/s3
        float yo = sy[n*3 + oc] * IS3;
        for (int v = 0; v < mi; ++v) pacc += (float)R[v] * (yo * f[v]);
      } else if (pe.lf == 0) {                  // lo=li=1, lf=0: f[v,o]/s3
        for (int v = 0; v < mi; ++v) pacc += (float)R[v] * (f[3*v + oc] * IS3);
      } else {                                  // lo=li=1, lf=1: eps/sqrt(6)
        int o1 = (oc + 1) % 3, o2 = (oc + 2) % 3;
        float ya = sy[n*3 + o1], yb = sy[n*3 + o2];
        for (int v = 0; v < mi; ++v) {
          float s = (yb * f[3*v + o1] - ya * f[3*v + o2]) * IS6;
          pacc += (float)R[v] * s;
        }
      }
    }
    acc += norm * pacc;
  }
  out[(size_t)(b * NPTS + a) * dout + t] = acc;
}

// ---------------------------------------------------------------------------
// Gated nonlinearity (layers 0..2)
// ---------------------------------------------------------------------------
__global__ void gate_kernel(int layer, const float* __restrict__ in,
                            float* __restrict__ out) {
  int blk = blockIdx.x, t = threadIdx.x;
  int dg = c_gDout[layer];
  if (t >= dg) return;
  int nscal = 16 * c_gNscalBlk[layer];
  int ngates = c_gNgates[layer];
  int dinL = c_dout[layer];
  const float* x = in + (size_t)blk * dinL;
  float o;
  if (t < nscal) {
    float v = x[t];
    o = (c_gPar[layer][t / 16] == 1) ? silu_f(v) : tanhf(v);
  } else {
    int tp = t - nscal;
    int k = tp / 48, uu = (tp % 48) / 3;
    float g = sigm_f(x[nscal + k * 16 + uu]);
    o = x[nscal + ngates + tp] * g;
  }
  out[(size_t)blk * dg + t] = o;
}

// ---------------------------------------------------------------------------
extern "C" void kernel_launch(void* const* d_in, const int* in_sizes, int n_in,
                              void* d_out, int out_size, void* d_ws, size_t ws_size,
                              hipStream_t stream) {
  (void)in_sizes; (void)n_in; (void)out_size; (void)ws_size;
  const float* features = (const float*)d_in[0];   // (2,96,4)
  const float* geometry = (const float*)d_in[1];   // (2,96,3)

  // workspace layout (bytes)
  char* base = (char*)d_ws;
  float*    Y1b    = (float*)   (base + 0);          //  18432*3*4
  float*    basisb = (float*)   (base + 221184);
  _Float16* H1     = (_Float16*)(base + 442368);     //  18432*128 f16
  _Float16* H2     = (_Float16*)(base + 5160960);
  _Float16* H3     = (_Float16*)(base + 9879552);
  _Float16* Wt     = (_Float16*)(base + 14598144);   //  3072*128 f16 (reused)
  float*    featA  = (float*)   (base + 15384576);   //  192*160 f32
  float*    featB  = (float*)   (base + 15507456);   //  192*128 f32
  _Float16* wbuf   = (_Float16*)(base + 15605760);   //  18432*3072 f16 (reused)

  const int NW[4] = {192, 2304, 3072, 256};
  const int Mtiles = PAIRS / 16;                     // 1152
  const int wavesPerBlk = 8;                         // 256 threads

  pair_kernel<<<(PAIRS + 255) / 256, 256, 0, stream>>>(geometry, Y1b, basisb);

  for (int L = 0; L < 4; ++L) {
    const float* W0 = (const float*)d_in[2 + L*8 + 0];
    const float* b0 = (const float*)d_in[2 + L*8 + 1];
    const float* W1 = (const float*)d_in[2 + L*8 + 2];
    const float* b1 = (const float*)d_in[2 + L*8 + 3];
    const float* W2 = (const float*)d_in[2 + L*8 + 4];
    const float* b2 = (const float*)d_in[2 + L*8 + 5];
    const float* W3 = (const float*)d_in[2 + L*8 + 6];
    const float* b3 = (const float*)d_in[2 + L*8 + 7];
    int nw = NW[L];

    // radial MLP: H1 (VALU, K=3) then two hidden + output GEMMs (WMMA)
    radial1_kernel<<<(PAIRS + 255) / 256, 256, 0, stream>>>(basisb, W0, b0, H1);

    convert_wt_kernel<<<(112 * KP + 255) / 256, 256, 0, stream>>>(W1, HID, HID, 112, Wt);
    {
      int T = Mtiles * 7;
      wmma_gemm_kernel<<<(T + wavesPerBlk - 1) / wavesPerBlk, 256, 0, stream>>>(
          H1, Wt, b1, Mtiles, 7, HID, 4, H2, KP, 1);
    }
    convert_wt_kernel<<<(112 * KP + 255) / 256, 256, 0, stream>>>(W2, HID, HID, 112, Wt);
    {
      int T = Mtiles * 7;
      wmma_gemm_kernel<<<(T + wavesPerBlk - 1) / wavesPerBlk, 256, 0, stream>>>(
          H2, Wt, b2, Mtiles, 7, HID, 4, H3, KP, 1);
    }
    convert_wt_kernel<<<(nw * KP + 255) / 256, 256, 0, stream>>>(W3, HID, nw, nw, Wt);
    {
      int Ntiles = nw / 16;
      int T = Mtiles * Ntiles;
      wmma_gemm_kernel<<<(T + wavesPerBlk - 1) / wavesPerBlk, 256, 0, stream>>>(
          H3, Wt, b3, Mtiles, Ntiles, nw, 4, wbuf, nw, 0);
    }

    // tensor-product conv + gate
    const float* featsrc = (L == 0) ? features : featB;
    float* convdst = (L == 3) ? (float*)d_out : featA;
    conv_kernel<<<NB_ * NPTS, 160, 0, stream>>>(L, featsrc, wbuf, nw, Y1b, convdst);
    if (L < 3)
      gate_kernel<<<NB_ * NPTS, 128, 0, stream>>>(L, featA, featB);
  }
}
